// GraphResBlock_57964878627089
// MI455X (gfx1250) — matrix operand order repaired and checked by reading, not compiled, not run on hardware
//
#include <hip/hip_runtime.h>
#include <hip/hip_bf16.h>

// ---------------------------------------------------------------------------
// GraphResBlock for MI455X (gfx1250, wave32, WMMA, async-to-LDS).
// deg==17 uniformly (every node has exactly K=16 in-edges + self loop), so the
// GCN aggregation is a deterministic 16-row gather with norm 1/17.
// All GEMM work uses v_wmma_f32_16x16x32_bf16 with a PRE-SPLIT bf16 hi/lo
// representation (dot ~= hi*hi + hi*lo + lo*hi): fp32-class accuracy at bf16
// matrix rates, zero conversion VALU in the hot loops.
// kNN kernel double-buffers candidate tiles in LDS via
// global_load_async_to_lds_b128 (ASYNCcnt) to overlap copy with WMMA.
// ---------------------------------------------------------------------------

#define NNODES   32768   // B*N
#define NG       16      // graphs
#define NPG      2048    // nodes per graph
#define DIN      128     // IN_DIM
#define DIM      256
#define KNN      16
#define NEDGE    (NNODES * KNN)

#define TQ 64            // query tile in knn kernel
#define TC 32            // candidate chunk in knn kernel

typedef __attribute__((ext_vector_type(16))) __bf16 v16bf;
typedef __attribute__((ext_vector_type(8)))  __bf16 v8bf;
typedef __attribute__((ext_vector_type(8)))  float  v8f;

#define ZERO8 {0.f, 0.f, 0.f, 0.f, 0.f, 0.f, 0.f, 0.f}

// --- CDNA5 async copy helpers (CDNA5 ISA) ----------------

__device__ __forceinline__ void async_cp16(const __bf16* __restrict__ gsrc,
                                           __bf16* lds_dst) {
  // LDS byte offset == low 32 bits of the generic pointer (ISA 10.2 aperture).
  unsigned l = (unsigned)(unsigned long long)lds_dst;
  asm volatile("global_load_async_to_lds_b128 %0, %1, off"
               :
               : "v"(l), "v"(gsrc)
               : "memory");
}

__device__ __forceinline__ void wait_async0() {
  asm volatile("s_wait_asynccnt 0" ::: "memory");
}

// --- WMMA helpers -----------------------------------------------------------

__device__ __forceinline__ void wmma3(const v16bf& ah, const v16bf& al,
                                      const v16bf& bh, const v16bf& bl,
                                      v8f& acc, v8f& acc2) {
  acc  = __builtin_amdgcn_wmma_f32_16x16x32_bf16(false, ah, false, bh, (short)0, acc,  false, false);
  acc2 = __builtin_amdgcn_wmma_f32_16x16x32_bf16(false, ah, false, bl, (short)0, acc2, false, false);
  acc2 = __builtin_amdgcn_wmma_f32_16x16x32_bf16(false, al, false, bh, (short)0, acc2, false, false);
}

// Load a v16bf fragment as two 16B vector loads at p and p+gap (elements).
// A-frag (16x32, ISA 7.12.2): per lane two contiguous runs of 8 -> gap=16.
// B-frag (32x16): same two runs of 8 as A                     -> gap=16.
__device__ __forceinline__ v16bf ld_gap(const __bf16* p, int gap) {
  v8bf a = *(const v8bf*)p;
  v8bf b = *(const v8bf*)(p + gap);
  return __builtin_shufflevector(a, b, 0, 1, 2, 3, 4, 5, 6, 7,
                                 8, 9, 10, 11, 12, 13, 14, 15);
}

__device__ __forceinline__ void bsplit(float v, __bf16& h, __bf16& l) {
  h = (__bf16)v;
  l = (__bf16)(v - (float)h);
}

// --- prep: split fp32 -> bf16 hi/lo -----------------------------------------

__global__ void split_x_kernel(const float* __restrict__ x,
                               __bf16* __restrict__ xh, __bf16* __restrict__ xl) {
  long i = (long)blockIdx.x * 256 + threadIdx.x;
  __bf16 h, l;
  bsplit(x[i], h, l);
  xh[i] = h; xl[i] = l;
}

// W [Kd x N] fp32 row-major -> transposed bf16 hi/lo [N x Kd]
__global__ void split_wT_kernel(const float* __restrict__ W,
                                __bf16* __restrict__ Wth, __bf16* __restrict__ Wtl,
                                int Kd, int Ncols) {
  long f = (long)blockIdx.x * 256 + threadIdx.x;   // f over N*Kd
  int col = (int)(f / Kd), k = (int)(f % Kd);
  __bf16 h, l;
  bsplit(W[(long)k * Ncols + col], h, l);
  Wth[f] = h; Wtl[f] = l;
}

// --- squared row norms ------------------------------------------------------

__global__ void sq_kernel(const float* __restrict__ x, float* __restrict__ sq) {
  int lane = threadIdx.x & 31, wave = threadIdx.x >> 5;
  long row = (long)blockIdx.x * 8 + wave;
  const float* p = x + row * DIN;
  float s = 0.f;
#pragma unroll
  for (int k = lane; k < DIN; k += 32) { float v = p[k]; s += v * v; }
#pragma unroll
  for (int off = 16; off >= 1; off >>= 1) s += __shfl_xor(s, off, 32);
  if (lane == 0) sq[row] = s;
}

// --- fused kNN: async double-buffered tiles + split-bf16 WMMA + top-16 ------

__global__ void __launch_bounds__(256)
knn_kernel(const __bf16* __restrict__ xh, const __bf16* __restrict__ xl,
           const float* __restrict__ sq,
           int* __restrict__ nbr, int* __restrict__ ei_src, int* __restrict__ ei_dst) {
  __shared__ __align__(16) __bf16 XcH[2][TC * DIN];   // 2 x 8KB
  __shared__ __align__(16) __bf16 XcL[2][TC * DIN];   // 2 x 8KB
  __shared__ float S[TQ * TC];                        // 8KB
  __shared__ float sqq[TQ];
  __shared__ float sqc[2][TC];

  int t = threadIdx.x, lane = t & 31, wave = t >> 5;
  int g  = blockIdx.x >> 5;               // 32 blocks per graph
  int q0 = (blockIdx.x & 31) * TQ;
  long gbase = (long)g * NPG;
  int rt = wave >> 1, ct = wave & 1;      // 8 waves <-> 4x2 tiles of 16x16

  // Query A-fragments are invariant over all candidate chunks: keep the whole
  // K=128 strip (4 k-chunks, hi+lo) in registers; no LDS for Xq at all.
  v16bf aH[4], aL[4];
  {
    long arow = gbase + q0 + rt * 16 + (lane & 15);
    const __bf16* pH = xh + arow * DIN + ((lane >> 4) << 3);
    const __bf16* pL = xl + arow * DIN + ((lane >> 4) << 3);
#pragma unroll
    for (int kc = 0; kc < 4; ++kc) {
      aH[kc] = ld_gap(pH + kc * 32, 16);
      aL[kc] = ld_gap(pL + kc * 32, 16);
    }
  }
  if (t < TQ) sqq[t] = sq[gbase + q0 + t];

  float bd[KNN]; int bi[KNN];
#pragma unroll
  for (int j = 0; j < KNN; ++j) { bd[j] = 3.0e38f; bi[j] = 0; }
  float wv = 3.0e38f; int wi = 0;

  // async-stage one candidate chunk (TC*DIN bf16 x2 = 2 b128 ops/thread/array)
  auto issue = [&](int bufi, int cb) {
    const __bf16* srcH = xh + (gbase + cb) * DIN;
    const __bf16* srcL = xl + (gbase + cb) * DIN;
    for (int f = t; f < TC * DIN / 8; f += 256) {
      async_cp16(srcH + f * 8, &XcH[bufi][f * 8]);
      async_cp16(srcL + f * 8, &XcL[bufi][f * 8]);
    }
    if (t < TC) sqc[bufi][t] = sq[gbase + cb + t];
  };

  issue(0, 0);
  int buf = 0;
  for (int cb = 0; cb < NPG; cb += TC, buf ^= 1) {
    wait_async0();       // my async copies for `buf` have landed in LDS
    __syncthreads();     // everyone's have
    if (cb + TC < NPG) issue(buf ^ 1, cb + TC);   // overlap next copy w/ WMMA

    v8f acc = ZERO8, acc2 = ZERO8;
    {
      const __bf16* pbH = &XcH[buf][(ct * 16 + (lane & 15)) * DIN + ((lane >> 4) << 3)];
      const __bf16* pbL = &XcL[buf][(ct * 16 + (lane & 15)) * DIN + ((lane >> 4) << 3)];
#pragma unroll
      for (int kc = 0; kc < 4; ++kc) {
        v16bf bh = ld_gap(pbH + kc * 32, 16);
        v16bf bl = ld_gap(pbL + kc * 32, 16);
        wmma3(aH[kc], aL[kc], bh, bl, acc, acc2);
      }
    }
    // C layout: VGPR r -> M = r + 8*(lane>=16), N = lane&15
    int n  = ct * 16 + (lane & 15);
    int m0 = rt * 16 + ((lane >> 4) << 3);
#pragma unroll
    for (int r = 0; r < 8; ++r) {
      int m = m0 + r;
      float d = sqq[m] + sqc[buf][n] - 2.0f * (acc[r] + acc2[r]);
      if (q0 + m == cb + n) d = 3.0e38f;   // exclude self (loop=False)
      S[m * TC + n] = d;
    }
    __syncthreads();

    if (t < TQ) {
      for (int n2 = 0; n2 < TC; ++n2) {
        float d = S[t * TC + n2];
        if (d < wv) {
          bd[wi] = d; bi[wi] = cb + n2;
          wv = bd[0]; wi = 0;
#pragma unroll
          for (int j = 1; j < KNN; ++j) if (bd[j] > wv) { wv = bd[j]; wi = j; }
        }
      }
    }
    __syncthreads();     // selection done before S / Xc[buf] are overwritten
  }

  if (t < TQ) {
    for (int a = 1; a < KNN; ++a) {        // ascending distance (top_k order)
      float dv = bd[a]; int iv = bi[a]; int p = a - 1;
      while (p >= 0 && bd[p] > dv) { bd[p + 1] = bd[p]; bi[p + 1] = bi[p]; --p; }
      bd[p + 1] = dv; bi[p + 1] = iv;
    }
    long qg = gbase + q0 + t;
#pragma unroll
    for (int j = 0; j < KNN; ++j) {
      int sgid = (int)gbase + bi[j];
      nbr[qg * KNN + j]    = sgid;
      ei_src[qg * KNN + j] = sgid;
      ei_dst[qg * KNN + j] = (int)qg;
    }
  }
}

// --- GEMM: C[M x N] = A @ W, pre-split bf16, W pre-transposed ---------------

__global__ void __launch_bounds__(256)
gemm_bf16_kernel(const __bf16* __restrict__ Ah, const __bf16* __restrict__ Al,
                 const __bf16* __restrict__ Wth, const __bf16* __restrict__ Wtl,
                 float* __restrict__ C, int Kd, int Ncols) {
  int lane = threadIdx.x & 31, wave = threadIdx.x >> 5;
  int col0 = blockIdx.x * 16;
  int row0 = (blockIdx.y * 8 + wave) * 16;
  const __bf16* pAh = Ah + (long)(row0 + (lane & 15)) * Kd + ((lane >> 4) << 3);
  const __bf16* pAl = Al + (long)(row0 + (lane & 15)) * Kd + ((lane >> 4) << 3);
  const __bf16* pBh = Wth + (long)(col0 + (lane & 15)) * Kd + ((lane >> 4) << 3);
  const __bf16* pBl = Wtl + (long)(col0 + (lane & 15)) * Kd + ((lane >> 4) << 3);
  v8f acc = ZERO8, acc2 = ZERO8;
  for (int kb = 0; kb < Kd; kb += 32) {
    v16bf ah = ld_gap(pAh + kb, 16);
    v16bf al = ld_gap(pAl + kb, 16);
    v16bf bh = ld_gap(pBh + kb, 16);
    v16bf bl = ld_gap(pBl + kb, 16);
    wmma3(ah, al, bh, bl, acc, acc2);
  }
  int n  = col0 + (lane & 15);
  int m0 = row0 + ((lane >> 4) << 3);
#pragma unroll
  for (int r = 0; r < 8; ++r)
    C[(long)(m0 + r) * Ncols + n] = acc[r] + acc2[r];
}

// --- H1 = relu((sum_nbr H0 + H0)/17 + b1), emitted as bf16 hi/lo ------------

__global__ void __launch_bounds__(256)
gather_relu_kernel(const float* __restrict__ H0, const int* __restrict__ nbr,
                   const float* __restrict__ b1,
                   __bf16* __restrict__ H1h, __bf16* __restrict__ H1l) {
  __shared__ int nb[KNN];
  long i = blockIdx.x;
  int c = threadIdx.x;
  if (c < KNN) nb[c] = nbr[i * KNN + c];
  __syncthreads();
#pragma unroll
  for (int j = 0; j < KNN; ++j)
    __builtin_prefetch(&H0[(long)nb[j] * DIM + c], 0, 0);   // global_prefetch_b8
  float s = H0[i * DIM + c];
#pragma unroll
  for (int j = 0; j < KNN; ++j) s += H0[(long)nb[j] * DIM + c];
  float v = s * (1.0f / 17.0f) + b1[c];
  v = v > 0.f ? v : 0.f;
  __bf16 h, l;
  bsplit(v, h, l);
  H1h[i * DIM + c] = h;
  H1l[i * DIM + c] = l;
}

// --- out = (sum_nbr G + G)/17 + b2 + x@Ws + bs ------------------------------

__global__ void __launch_bounds__(256)
final_kernel(const __bf16* __restrict__ xh, const __bf16* __restrict__ xl,
             const __bf16* __restrict__ Wsth, const __bf16* __restrict__ Wstl,
             const float* __restrict__ bs, const float* __restrict__ b2,
             const float* __restrict__ G, const int* __restrict__ nbr,
             float* __restrict__ out) {
  __shared__ float sc[16 * DIM];     // 16KB shortcut tile
  __shared__ int   nb[16 * KNN];
  int t = threadIdx.x, lane = t & 31, wave = t >> 5;
  int row0 = blockIdx.x * 16;

  // shortcut GEMM x@Ws: 16 rows x 256 cols; A frags hoisted, 2 col-tiles/wave
  v16bf aH[4], aL[4];
  {
    const __bf16* pAh = xh + (long)(row0 + (lane & 15)) * DIN + ((lane >> 4) << 3);
    const __bf16* pAl = xl + (long)(row0 + (lane & 15)) * DIN + ((lane >> 4) << 3);
#pragma unroll
    for (int kc = 0; kc < 4; ++kc) {
      aH[kc] = ld_gap(pAh + kc * 32, 16);
      aL[kc] = ld_gap(pAl + kc * 32, 16);
    }
  }
#pragma unroll
  for (int ci = 0; ci < 2; ++ci) {
    int col0 = (wave * 2 + ci) * 16;
    const __bf16* pBh = Wsth + (long)(col0 + (lane & 15)) * DIN + ((lane >> 4) << 3);
    const __bf16* pBl = Wstl + (long)(col0 + (lane & 15)) * DIN + ((lane >> 4) << 3);
    v8f acc = ZERO8, acc2 = ZERO8;
#pragma unroll
    for (int kc = 0; kc < 4; ++kc) {
      v16bf bh = ld_gap(pBh + kc * 32, 16);
      v16bf bl = ld_gap(pBl + kc * 32, 16);
      wmma3(aH[kc], aL[kc], bh, bl, acc, acc2);
    }
    int n  = col0 + (lane & 15);
    int m0 = (lane >> 4) << 3;
#pragma unroll
    for (int r = 0; r < 8; ++r) sc[(m0 + r) * DIM + n] = acc[r] + acc2[r];
  }
  nb[t] = nbr[(long)(row0 + (t >> 4)) * KNN + (t & 15)];
  __syncthreads();

  int c = t;
  float bias = b2[c] + bs[c];
  for (int m = 0; m < 16; ++m) {
    long i = row0 + m;
#pragma unroll
    for (int j = 0; j < KNN; ++j)
      __builtin_prefetch(&G[(long)nb[m * KNN + j] * DIM + c], 0, 0);
    float s = G[i * DIM + c];
#pragma unroll
    for (int j = 0; j < KNN; ++j) s += G[(long)nb[m * KNN + j] * DIM + c];
    out[i * DIM + c] = s * (1.0f / 17.0f) + bias + sc[m * DIM + c];
  }
}

// --- launch -----------------------------------------------------------------

extern "C" void kernel_launch(void* const* d_in, const int* in_sizes, int n_in,
                              void* d_out, int out_size, void* d_ws, size_t ws_size,
                              hipStream_t stream) {
  const float* x  = (const float*)d_in[0];
  // d_in[1] = batch (graph membership is implied by node index)
  const float* W1 = (const float*)d_in[2];
  const float* b1 = (const float*)d_in[3];
  const float* W2 = (const float*)d_in[4];
  const float* b2 = (const float*)d_in[5];
  const float* Ws = (const float*)d_in[6];
  const float* bs = (const float*)d_in[7];

  float* out    = (float*)d_out;
  int*   ei     = (int*)(out + (long)NNODES * DIM);   // edge_index tail
  int*   ei_src = ei;
  int*   ei_dst = ei + NEDGE;

  // workspace layout (~115 MB, all 16B-aligned regions)
  char* w = (char*)d_ws;
  float*  sq   = (float*)w;    w += (long)NNODES * 4;
  int*    nbr  = (int*)w;      w += (long)NNODES * KNN * 4;
  __bf16* xh   = (__bf16*)w;   w += (long)NNODES * DIN * 2;
  __bf16* xl   = (__bf16*)w;   w += (long)NNODES * DIN * 2;
  __bf16* W1h  = (__bf16*)w;   w += (long)DIN * DIM * 2;
  __bf16* W1l  = (__bf16*)w;   w += (long)DIN * DIM * 2;
  __bf16* W2h  = (__bf16*)w;   w += (long)DIM * DIM * 2;
  __bf16* W2l  = (__bf16*)w;   w += (long)DIM * DIM * 2;
  __bf16* Wsh  = (__bf16*)w;   w += (long)DIN * DIM * 2;
  __bf16* Wsl  = (__bf16*)w;   w += (long)DIN * DIM * 2;
  float*  H0   = (float*)w;    w += (long)NNODES * DIM * 4;
  __bf16* H1h  = (__bf16*)w;   w += (long)NNODES * DIM * 2;
  __bf16* H1l  = (__bf16*)w;   w += (long)NNODES * DIM * 2;
  float*  G    = (float*)w;

  split_x_kernel<<<NNODES * DIN / 256, 256, 0, stream>>>(x, xh, xl);
  split_wT_kernel<<<DIN * DIM / 256, 256, 0, stream>>>(W1, W1h, W1l, DIN, DIM);
  split_wT_kernel<<<DIM * DIM / 256, 256, 0, stream>>>(W2, W2h, W2l, DIM, DIM);
  split_wT_kernel<<<DIN * DIM / 256, 256, 0, stream>>>(Ws, Wsh, Wsl, DIN, DIM);
  sq_kernel<<<NNODES / 8, 256, 0, stream>>>(x, sq);
  knn_kernel<<<NG * (NPG / TQ), 256, 0, stream>>>(xh, xl, sq, nbr, ei_src, ei_dst);
  gemm_bf16_kernel<<<dim3(DIM / 16, NNODES / 128), 256, 0, stream>>>(xh, xl, W1h, W1l, H0, DIN, DIM);
  gather_relu_kernel<<<NNODES, 256, 0, stream>>>(H0, nbr, b1, H1h, H1l);
  gemm_bf16_kernel<<<dim3(DIM / 16, NNODES / 128), 256, 0, stream>>>(H1h, H1l, W2h, W2l, G, DIM, DIM);
  final_kernel<<<NNODES / 16, 256, 0, stream>>>(xh, xl, Wsh, Wsl, bs, b2, G, nbr, out);
}
